// Context_atten_3255585210564
// MI455X (gfx1250) — matrix operand order repaired and checked by reading, not compiled, not run on hardware
//
#include <hip/hip_runtime.h>

// Problem constants (B, S, H, W) = (256, 512, 1024, 64)
#define H       1024
#define SEQ     512
#define NB      256
#define WW      64
#define THREADS 512
#define CHUNK   64                  // output columns per LDS panel
#define NCHUNKS (H / CHUNK)         // 16
#define PANEL_ELEMS (H * CHUNK)     // 65536 bf16 per chunk panel
#define MT_STRIDE 1032              // bf16 per Mt row (1024 + 8 pad, 16B aligned rows)
#define B_STRIDE  72                // bf16 per B-panel row (64 + 8 pad -> 144B rows, 16B aligned)

typedef __bf16 bf16_t;
typedef __bf16 v16bf __attribute__((ext_vector_type(16)));
typedef float  v8f   __attribute__((ext_vector_type(8)));
typedef unsigned int u32x4 __attribute__((ext_vector_type(4)));
typedef unsigned int u32x2 __attribute__((ext_vector_type(2)));
typedef float  f32x4 __attribute__((ext_vector_type(4)));

// ---- dynamic LDS layout (bytes); no static __shared__ anywhere, so dynamic
// LDS starts at segment offset 0 and these constants ARE the LDS addresses. ----
#define OFF_MT 0
#define SZ_MT  (WW * MT_STRIDE * 2)       // 132096
#define OFF_B  (OFF_MT + SZ_MT)           // 132096
#define SZ_B   (H * B_STRIDE * 2)         // 147456
#define OFF_HB (OFF_B + SZ_B)             // 279552
#define OFF_V  (OFF_HB + H * 4)           // 283648
#define OFF_SC (OFF_V + H * 4)            // 287744
#define OFF_AT (OFF_SC + WW * 4)          // 288000
#define LDS_MAIN (OFF_AT + WW * 4)        // 288256  (< 320KB WGP LDS)
#define LDS_HB   OFF_HB                    // 279552 for the hb GEMM kernel

// ---------------- helpers ----------------
__device__ __forceinline__ unsigned short f2bf_bits(float f) {
  unsigned u = __float_as_uint(f);
  u += 0x7FFFu + ((u >> 16) & 1u);        // round-to-nearest-even
  return (unsigned short)(u >> 16);
}
__device__ __forceinline__ unsigned pack_bf2(float lo, float hi) {
  return (unsigned)f2bf_bits(lo) | ((unsigned)f2bf_bits(hi) << 16);
}
__device__ __forceinline__ float bf2f(bf16_t b) {
  unsigned short s = __builtin_bit_cast(unsigned short, b);
  return __uint_as_float((unsigned)s << 16);
}

__device__ __forceinline__ void wait_async0() {
#if __has_builtin(__builtin_amdgcn_s_wait_asynccnt)
  __builtin_amdgcn_s_wait_asynccnt(0);
#else
  asm volatile("s_wait_asynccnt 0x0" ::: "memory");
#endif
}

// Stage 64 rows x 1024 f32 from global into bf16 LDS [64][MT_STRIDE]
// (needs f32->bf16 conversion, so it goes through VGPRs)
__device__ __forceinline__ void stage_rows_bf16(bf16_t* sDst, const float* src,
                                                long rowStride) {
  const int tid = threadIdx.x;
#pragma unroll
  for (int t = 0; t < 32; ++t) {
    int idx = t * THREADS + tid;          // 16384 float4 groups
    int r   = idx >> 8;                   // row 0..63
    int cq  = idx & 255;                  // float4 within row
    f32x4 f = *(const f32x4*)(src + (long)r * rowStride + cq * 4);
    u32x2 p; p.x = pack_bf2(f.x, f.y); p.y = pack_bf2(f.z, f.w);
    *(u32x2*)(sDst + r * MT_STRIDE + cq * 4) = p;
  }
}

// Stage one contiguous bf16 panel (1024 x 64) straight into LDS [1024][B_STRIDE]
// using CDNA5 async global->LDS DMA (ASYNCcnt, no VGPR roundtrip).
// ldsBase: byte address of panel start inside the LDS segment.
__device__ __forceinline__ void stage_panel_async(unsigned ldsBase,
                                                  const bf16_t* src) {
  const int tid = threadIdx.x;
  const unsigned long long gbase = (unsigned long long)(uintptr_t)src;
#pragma unroll
  for (int t = 0; t < 16; ++t) {
    int idx = t * THREADS + tid;          // 8192 groups of 16B
    int k = idx >> 3;                     // panel row (K)
    int g = idx & 7;                      // 16B group within row
    unsigned ldsOff = ldsBase + (unsigned)(k * (B_STRIDE * 2) + g * 16);
    unsigned gOff   = (unsigned)idx * 16u;
    // GLOBAL_LOAD_ASYNC_TO_LDS_B128: vdst = per-lane LDS byte address,
    // vaddr = 32-bit offset, saddr = 64-bit base (GVS mode).
    asm volatile("global_load_async_to_lds_b128 %0, %1, %2"
                 :: "v"(ldsOff), "v"(gOff), "s"(gbase)
                 : "memory");
  }
}

// One 16x16 output tile: A rows from sA (16-bit A 16x32 layout: lanes 0-15 hold
// M=lane with K {0..7,16..23}; lanes 16-31 hold M=lane-16 with K {8..15,24..31}),
// B from sB [k][n] (lane -> N col = lane%16, same K order as A).
__device__ __forceinline__ v8f gemm_tile(const bf16_t* sA, const bf16_t* sB,
                                         int mrow, int aK0, int aK1,
                                         int lane, int nt) {
  v8f acc = {};
  const bf16_t* aRow = sA + mrow * MT_STRIDE;
#pragma unroll 4
  for (int k0 = 0; k0 < H; k0 += 32) {
    union { u32x4 q[2]; v16bf v; } A, Bm;
    A.q[0] = *(const u32x4*)(aRow + k0 + aK0);
    A.q[1] = *(const u32x4*)(aRow + k0 + aK1);
    const bf16_t* bCol = sB + k0 * B_STRIDE + nt * 16 + (lane & 15);
#pragma unroll
    for (int s = 0; s < 8; ++s) {
      Bm.v[s]     = bCol[((lane >> 4) * 8 + s) * B_STRIDE];
      Bm.v[8 + s] = bCol[(16 + (lane >> 4) * 8 + s) * B_STRIDE];
    }
    acc = __builtin_amdgcn_wmma_f32_16x16x32_bf16(false, A.v, false, Bm.v,
                                                  (short)0, acc, false, false);
  }
  return acc;
}

// ---------------- kernel 0: weight transpose/convert/swizzle ----------------
// W1sw/W2sw layout: [chunk c][k 0..1023][n 0..63], bf16, each panel contiguous.
// B[k][n] = W_attn[n_global][k (+1024 for W2)]
__global__ void prep_w(const float* __restrict__ W_attn,
                       bf16_t* __restrict__ W1sw, bf16_t* __restrict__ W2sw) {
  long idx = (long)blockIdx.x * blockDim.x + threadIdx.x;
  if (idx >= (long)H * H) return;
  int nl = (int)(idx & 63);
  int k  = (int)((idx >> 6) & (H - 1));
  int c  = (int)(idx >> 16);
  long row = (long)(c * 64 + nl) * (2 * H);
  W1sw[idx] = __builtin_bit_cast(bf16_t, f2bf_bits(W_attn[row + k]));
  W2sw[idx] = __builtin_bit_cast(bf16_t, f2bf_bits(W_attn[row + H + k]));
}

// ---------------- kernel 1: hb = hc @ W1^T + b_attn  (256x1024x1024) --------
__global__ __launch_bounds__(THREADS)
void hb_gemm(const float* __restrict__ hc, const float* __restrict__ b_attn,
             const bf16_t* __restrict__ W1sw, float* __restrict__ hb) {
  extern __shared__ char smem[];
  bf16_t* sA = (bf16_t*)(smem + OFF_MT);
  bf16_t* sB = (bf16_t*)(smem + OFF_B);

  const int mb = blockIdx.x >> 4;   // 0..3  -> rows mb*64..
  const int c  = blockIdx.x & 15;   // 0..15 -> cols c*64..

  stage_panel_async(OFF_B, W1sw + (long)c * PANEL_ELEMS);   // DMA, overlaps below
  stage_rows_bf16(sA, hc + (long)mb * 64 * H, H);           // VALU convert path
  wait_async0();
  __syncthreads();

  const int tid  = threadIdx.x;
  const int wave = tid >> 5, lane = tid & 31;
  const int m0 = (wave & 3) * 16, nt = wave >> 2;
  const int half = lane >> 4, ncol = lane & 15;
  const int mrow = m0 + ncol;
  const int aK0 = half * 8, aK1 = 16 + half * 8;

  v8f acc = gemm_tile(sA, sB, mrow, aK0, aK1, lane, nt);

  const int ng = c * CHUNK + nt * 16 + ncol;
  const float bias = b_attn[ng];
#pragma unroll
  for (int i = 0; i < 8; ++i) {
    int rg = mb * 64 + m0 + half * 8 + i;
    hb[(long)rg * H + ng] = acc[i] + bias;
  }
}

// ---------------- kernel 2: fused attention (one block per batch) -----------
__global__ __launch_bounds__(THREADS)
void ctx_attn_main(const float* __restrict__ inputs, const float* __restrict__ v,
                   const int* __restrict__ slen, const int* __restrict__ parent,
                   const bf16_t* __restrict__ W2sw, const float* __restrict__ hb,
                   float* __restrict__ out_attn, float* __restrict__ out_ctx,
                   float* __restrict__ out_par) {
  extern __shared__ char smem[];
  bf16_t* sMt = (bf16_t*)(smem + OFF_MT);
  bf16_t* sB  = (bf16_t*)(smem + OFF_B);
  float*  sHb = (float*)(smem + OFF_HB);
  float*  sV  = (float*)(smem + OFF_V);
  float*  sSc = (float*)(smem + OFF_SC);
  float*  sAt = (float*)(smem + OFF_AT);

  const int b   = blockIdx.x;
  const int tid = threadIdx.x;
  const int sl  = slen[b];
  const long inBase = (long)b * SEQ * H;

  if (tid < WW) sSc[tid] = 0.0f;
  for (int i = tid; i < H; i += THREADS) { sHb[i] = hb[(long)b * H + i]; sV[i] = v[i]; }

  // gather Mt window (rows sl-64 .. sl-1) into bf16 LDS
  stage_rows_bf16(sMt, inputs + inBase + (long)(sl - WW) * H, H);

  // par output: exact f32 row copy (independent of LDS)
  {
    const float* src = inputs + inBase + (long)(sl - parent[b] - 1) * H;
    float* dst = out_par + (long)b * H;
    for (int i = tid; i < H; i += THREADS) dst[i] = src[i];
  }

  const int wave = tid >> 5, lane = tid & 31;
  const int m0 = (wave & 3) * 16, nt = wave >> 2;
  const int half = lane >> 4, ncol = lane & 15;
  const int mrow = m0 + ncol;
  const int aK0 = half * 8, aK1 = 16 + half * 8;

  for (int c = 0; c < NCHUNKS; ++c) {
    __syncthreads();                       // all readers of previous sB done
    stage_panel_async(OFF_B, W2sw + (long)c * PANEL_ELEMS);  // DMA panel
    if (c + 1 < NCHUNKS) {                 // pull next panel toward GL2
      const bf16_t* nx = W2sw + (long)(c + 1) * PANEL_ELEMS;
      __builtin_prefetch(nx + (long)tid * 128, 0, 1);
      __builtin_prefetch(nx + (long)(tid + THREADS) * 128, 0, 1);
    }
    wait_async0();                         // this lane's async writes landed
    __syncthreads();                       // everyone's panel writes visible

    v8f acc = gemm_tile(sMt, sB, mrow, aK0, aK1, lane, nt);

    // fused epilogue: scores[m] += sum_n v[n] * tanh(E[m][n] + hb[n])
    const int ng  = c * CHUNK + nt * 16 + ncol;
    const float hbv = sHb[ng];
    const float vv  = sV[ng];
#pragma unroll
    for (int i = 0; i < 8; ++i) {
      float p = vv * tanhf(acc[i] + hbv);
      p += __shfl_xor(p, 1);
      p += __shfl_xor(p, 2);
      p += __shfl_xor(p, 4);
      p += __shfl_xor(p, 8);               // row-sum over the 16 N lanes
      if (ncol == 0) atomicAdd(&sSc[m0 + half * 8 + i], p);
    }
  }
  __syncthreads();

  // softmax over 64 scores (trivial cost vs GEMM)
  if (tid == 0) {
    float mx = sSc[0];
    for (int w2 = 1; w2 < WW; ++w2) mx = fmaxf(mx, sSc[w2]);
    float sum = 0.0f;
    for (int w2 = 0; w2 < WW; ++w2) { float e = __expf(sSc[w2] - mx); sAt[w2] = e; sum += e; }
    float inv = 1.0f / sum;
    for (int w2 = 0; w2 < WW; ++w2) sAt[w2] *= inv;
  }
  __syncthreads();

  if (tid < WW) out_attn[(long)b * WW + tid] = sAt[tid];

  // context[o] = sum_w attn[w] * Mt[w][o]   (Mt from LDS)
  for (int o = tid; o < H; o += THREADS) {
    float s = 0.0f;
#pragma unroll 8
    for (int w2 = 0; w2 < WW; ++w2) s += sAt[w2] * bf2f(sMt[w2 * MT_STRIDE + o]);
    out_ctx[(long)b * H + o] = s;
  }
}

// ---------------- host launcher ----------------
extern "C" void kernel_launch(void* const* d_in, const int* in_sizes, int n_in,
                              void* d_out, int out_size, void* d_ws, size_t ws_size,
                              hipStream_t stream) {
  const float* inputs = (const float*)d_in[0];
  const float* hc     = (const float*)d_in[1];
  const float* W_attn = (const float*)d_in[2];
  const float* b_attn = (const float*)d_in[3];
  const float* v      = (const float*)d_in[4];
  const int*   slen   = (const int*)d_in[5];
  const int*   parent = (const int*)d_in[6];

  float* out      = (float*)d_out;
  float* out_attn = out;                        // (B,1,W)  = 16384
  float* out_ctx  = out + (long)NB * WW;        // (B,1,H)  = 262144
  float* out_par  = out + (long)NB * WW + (long)NB * H;  // (B,H)

  char* ws = (char*)d_ws;
  bf16_t* W1sw = (bf16_t*)(ws);                             // 2 MB
  bf16_t* W2sw = (bf16_t*)(ws + (size_t)2 * 1024 * 1024);   // 2 MB
  float*  hb   = (float*)(ws + (size_t)4 * 1024 * 1024);    // 1 MB

  prep_w<<<(H * H) / 256, 256, 0, stream>>>(W_attn, W1sw, W2sw);
  hb_gemm<<<64, THREADS, LDS_HB, stream>>>(hc, b_attn, W1sw, hb);
  ctx_attn_main<<<NB, THREADS, LDS_MAIN, stream>>>(inputs, v, slen, parent,
                                                   W2sw, hb,
                                                   out_attn, out_ctx, out_par);
}
